// StyledLayer_42176578847438
// MI455X (gfx1250) — matrix operand order repaired and checked by reading, not compiled, not run on hardware
//
#include <hip/hip_runtime.h>
#include <hip/hip_bf16.h>

typedef __attribute__((ext_vector_type(16))) __bf16 v16bf;
typedef __attribute__((ext_vector_type(8)))  __bf16 v8bf;
typedef __attribute__((ext_vector_type(8)))  float  v8f;

union Frag16 { v16bf v; v8bf h[2]; };

#define B_    8
#define CIN_  64
#define COUT_ 64
#define S_    512
#define H_    256
#define W_    256

// ---------------------------------------------------------------------------
// Kernel 1: per-sample style scales  y[b][ci] = (w[b,:] . lin_w[ci,:])/sqrt(S) + lin_b[ci]
// ---------------------------------------------------------------------------
__global__ void k_style(const float* __restrict__ w, const float* __restrict__ lin_w,
                        const float* __restrict__ lin_b, float* __restrict__ y) {
    const int b  = blockIdx.x;   // 8
    const int ci = threadIdx.x;  // 64
    const float scale = 0.044194173824159216f; // 1/sqrt(512)
    const float* wr = w + b * S_;
    const float* lr = lin_w + ci * S_;
    float acc = 0.f;
    for (int s = 0; s < S_; ++s) acc += wr[s] * lr[s];
    y[b * CIN_ + ci] = acc * scale + lin_b[ci];
}

// ---------------------------------------------------------------------------
// Kernel 2: modulate + demodulate, emit bf16 weights in layout A[b][co][tap][ci]
// ---------------------------------------------------------------------------
__global__ void k_modw(const float* __restrict__ conv_w, const float* __restrict__ y,
                       __bf16* __restrict__ A) {
    const int b  = blockIdx.x >> 6;   // grid = B*COUT
    const int co = blockIdx.x & 63;
    const int ci = threadIdx.x;       // 64
    const float conv_scale = 1.0f / 24.0f; // 1/sqrt(64*9)
    const float s = conv_scale * y[b * CIN_ + ci];
    const float* cw = conv_w + (co * CIN_ + ci) * 9;
    float v[9], ss = 0.f;
#pragma unroll
    for (int j = 0; j < 9; ++j) { v[j] = cw[j] * s; ss += v[j] * v[j]; }

    __shared__ float red[64];
    red[ci] = ss;
    __syncthreads();
    for (int off = 32; off > 0; off >>= 1) {
        if (ci < off) red[ci] += red[ci + off];
        __syncthreads();
    }
    const float demod = rsqrtf(red[0] + 1e-8f);
    __bf16* outp = A + ((b * COUT_ + co) * 9) * CIN_ + ci;
#pragma unroll
    for (int j = 0; j < 9; ++j) outp[j * CIN_] = (__bf16)(v[j] * demod);
}

// ---------------------------------------------------------------------------
// Kernel 3: implicit-GEMM conv row. One block = (batch b, output row h).
//   M = Cout = 64 (4 mtiles), N = W = 256 (16 ntiles), K = 9 taps x 64 ci.
//   8 waves: wave -> (mtile = wave&3, N-half = wave>>2), 8 accum tiles each.
//   LDS: 3 halo rows x 258 px x 32-ci chunk, bf16, ci-minor (pad 32->40:
//   px stride 80B, 16B-aligned fragments, conflict-free dword stride 20).
//   Staging uses float4 global loads (16B per thread per iteration).
// ---------------------------------------------------------------------------
__global__ __launch_bounds__(256)
void k_conv(const float* __restrict__ x, const float* __restrict__ noise,
            const __bf16* __restrict__ A, const float* __restrict__ act_bias,
            const float* __restrict__ nwp, float* __restrict__ out) {
    const int b    = blockIdx.x >> 8;
    const int h    = blockIdx.x & 255;
    const int tid  = threadIdx.x;
    const int lane = tid & 31;
    const int wave = tid >> 5;
    const int mtile = wave & 3;     // which 16 rows of Cout
    const int nhalf = wave >> 2;    // which 128 pixels
    const int ln16  = lane & 15;
    const int hi    = lane >> 4;    // lane group (0: lanes 0-15, 1: lanes 16-31)

    __shared__ __align__(16) __bf16 lds[3][258][40];   // 61,920 B

    v8f acc[8];
    const v8f vzero = {0.f, 0.f, 0.f, 0.f, 0.f, 0.f, 0.f, 0.f};
#pragma unroll
    for (int i = 0; i < 8; ++i) acc[i] = vzero;

    const int co0 = mtile * 16 + ln16;      // A-fragment row for this lane

    for (int chunk = 0; chunk < 2; ++chunk) {
        __syncthreads();   // protect LDS from previous chunk's readers

        // Stage 3 halo rows x 32 channels x 256 px (fp32 -> bf16, transposed).
        // Each thread loads a float4 (consecutive threads -> consecutive 16B
        // segments of a row), converts, and scatters 4 bf16 into the
        // ci-minor LDS tile. 3*32*64 = 6144 float4 items / 256 thr = 24 iters.
#pragma unroll 4
        for (int it = 0; it < 24; ++it) {
            const int idx = tid + it * 256;
            const int w4 = idx & 63;           // which 4-px group in the row
            const int ci = (idx >> 6) & 31;
            const int kh = idx >> 11;          // /(64*32)
            const int hh = h + kh - 1;
            float4 val = make_float4(0.f, 0.f, 0.f, 0.f);
            if (hh >= 0 && hh < H_) {
                const float4* src = (const float4*)(x +
                    (((size_t)(b * CIN_ + chunk * 32 + ci) * H_) + hh) * W_);
                val = src[w4];
            }
            const int w = w4 * 4;
            lds[kh][w + 1][ci] = (__bf16)val.x;
            lds[kh][w + 2][ci] = (__bf16)val.y;
            lds[kh][w + 3][ci] = (__bf16)val.z;
            lds[kh][w + 4][ci] = (__bf16)val.w;
        }
        // Zero the left/right padding columns (px = 0 and px = 257).
        if (tid < 192) {
            const int ci   = tid & 31;
            const int side = (tid >> 5) & 1;
            const int kh   = tid >> 6;
            lds[kh][side ? 257 : 0][ci] = (__bf16)0.f;
        }
        __syncthreads();

        for (int tap = 0; tap < 9; ++tap) {
            const int kh = tap / 3;
            const int kw = tap - kh * 3;

            // A fragment: 16x32 bf16; lane row = co0, K contiguous 16*hi..16*hi+15.
            Frag16 a;
            const v8bf* pa = (const v8bf*)(A + (((b * COUT_ + co0) * 9 + tap) * CIN_
                                                + chunk * 32 + hi * 16));
            a.h[0] = pa[0];
            a.h[1] = pa[1];    // +8 bf16 elements

            const int cib = hi * 16;   // B fragment K base within 32-ci chunk
#pragma unroll
            for (int nt = 0; nt < 8; ++nt) {
                const int px = nhalf * 128 + nt * 16 + ln16 + kw; // +1 pad, -1 shift folded
                Frag16 bf;
                const v8bf* pb = (const v8bf*)(&lds[kh][px][cib]);
                bf.h[0] = pb[0];
                bf.h[1] = pb[1];
                acc[nt] = __builtin_amdgcn_wmma_f32_16x16x32_bf16(
                    false, a.v, false, bf.v, (short)0, acc[nt], false, false);
            }
        }
    }

    // Epilogue: noise injection + bias + leaky_relu(0.2) * sqrt(2)
    const float nw = nwp[0];
    const float* nrow = noise + (b * H_ + h) * W_;
#pragma unroll
    for (int nt = 0; nt < 8; ++nt) {
        const int px = nhalf * 128 + nt * 16 + ln16;
        const float nz = nw * nrow[px];
#pragma unroll
        for (int r = 0; r < 8; ++r) {
            const int co = mtile * 16 + hi * 8 + r;   // C/D layout: M = r + 8*hi
            float v = acc[nt][r] + nz + act_bias[co];
            v = (v > 0.f ? v : 0.2f * v) * 1.41421356237309515f;
            out[(((size_t)(b * COUT_ + co) * H_) + h) * W_ + px] = v;
        }
    }
}

// ---------------------------------------------------------------------------
extern "C" void kernel_launch(void* const* d_in, const int* in_sizes, int n_in,
                              void* d_out, int out_size, void* d_ws, size_t ws_size,
                              hipStream_t stream) {
    const float* x        = (const float*)d_in[0];
    const float* w        = (const float*)d_in[1];
    const float* noise    = (const float*)d_in[2];
    const float* lin_w    = (const float*)d_in[3];
    const float* lin_b    = (const float*)d_in[4];
    const float* conv_w   = (const float*)d_in[5];
    const float* noise_wt = (const float*)d_in[6];
    const float* act_bias = (const float*)d_in[7];
    float* outp = (float*)d_out;

    // Workspace: bf16 modulated weights (576 KB) + fp32 styles (2 KB)
    __bf16* A = (__bf16*)d_ws;
    float*  y = (float*)((char*)d_ws + (size_t)B_ * COUT_ * 9 * CIN_ * sizeof(__bf16));

    k_style<<<B_, CIN_, 0, stream>>>(w, lin_w, lin_b, y);
    k_modw <<<B_ * COUT_, CIN_, 0, stream>>>(conv_w, y, A);
    k_conv <<<B_ * H_, 256, 0, stream>>>(x, noise, A, act_bias, noise_wt, outp);
}
